// TTS_45681272160861
// MI455X (gfx1250) — matrix operand order repaired and checked by reading, not compiled, not run on hardware
//
#include <hip/hip_runtime.h>

// ---------------------------------------------------------------------------
// TTS GMM-attention LSTM recurrence, persistent-kernel implementation.
// B=32, T=512, D=1024, S=256, M=10. K_total = 3072 (= inp 1024 | ctx 1024 | h 1024)
// gates GEMM: [32,3072] x [3072,4096] in bf16 WMMA (f32 accumulate).
// W rows permuted so output col n = 4*d + gate  ->  one 16-col tile = 4 channels
// x 4 gates  ->  LSTM cell update fused into the GEMM wave.
// ---------------------------------------------------------------------------

typedef __attribute__((ext_vector_type(16))) __bf16 v16bf;
typedef __attribute__((ext_vector_type(8)))  __bf16 v8bf;
typedef __attribute__((ext_vector_type(8)))  float  v8f;

#define NBLK   64
#define BTHR   256
#define BD     32          // batch
#define TT     512         // time steps
#define DD     1024        // hidden
#define SS     256         // memory length
#define MM     10          // mixtures
#define KTOT   3072
#define NTOT   4096
#define KTILES 96          // KTOT/32
#define NTILES 256         // NTOT/16

__device__ __forceinline__ __bf16 f2bf(float f) {
    union { float f; unsigned u; } x; x.f = f;
    unsigned u = x.u;
    unsigned r = u + 0x7FFFu + ((u >> 16) & 1u);   // round to nearest even
    unsigned short s = (unsigned short)(r >> 16);
    union { unsigned short s; __bf16 b; } y; y.s = s; return y.b;
}

__device__ __forceinline__ float sigf(float x) { return 1.0f / (1.0f + __expf(-x)); }

// one 16x16x32 bf16 WMMA step; arow points at this lane's first 8 K-elements,
// second group of 8 lives 16 elements further (ISA 16-bit A layout).
__device__ __forceinline__ v8f mma32(const __bf16* arow, const __bf16* bp, v8f acc) {
    union { v16bf v; v8bf h[2]; } a;
    a.h[0] = *(const v8bf*)(arow);
    a.h[1] = *(const v8bf*)(arow + 16);
    v16bf b = *(const v16bf*)bp;
    return __builtin_amdgcn_wmma_f32_16x16x32_bf16(false, a.v, false, b,
                                                   (short)0, acc, false, false);
}

// device-wide ticket barrier (all NBLK blocks resident; counter monotonic)
__device__ __forceinline__ void grid_barrier(unsigned* bar) {
    __syncthreads();
    if (threadIdx.x == 0) {
        __threadfence();
        unsigned ticket = __hip_atomic_fetch_add(bar, 1u, __ATOMIC_ACQ_REL,
                                                 __HIP_MEMORY_SCOPE_AGENT);
        unsigned target = (ticket / NBLK + 1u) * NBLK;
        while (__hip_atomic_load(bar, __ATOMIC_ACQUIRE,
                                 __HIP_MEMORY_SCOPE_AGENT) < target) {
            __builtin_amdgcn_s_sleep(2);
        }
    }
    __syncthreads();
}

// ---------------------------------------------------------------------------
// One-time prep kernels
// ---------------------------------------------------------------------------

__global__ void init_state(float* c, float* h, __bf16* hb, __bf16* ctxb,
                           float* ksi, unsigned* bar) {
    int i = blockIdx.x * blockDim.x + threadIdx.x;
    if (i < BD * DD) { c[i] = 0.f; h[i] = 0.f; hb[i] = f2bf(0.f); ctxb[i] = f2bf(0.f); }
    if (i < BD * MM) ksi[i] = 0.f;
    if (i == 0) *bar = 0u;
}

// Pack W_cat = [W_ih | W_hh] (rows permuted n=4d+gate) into WMMA-B fragment
// order: idx = ((nt*96 + kt)*32 + lane)*16 + e; lane = col(0..15) + 16*khalf,
// fragment element e -> K = (e&7) + 16*(e>>3) + 8*khalf within the 32-wide K tile.
__global__ void pack_w(const float* __restrict__ W_ih, const float* __restrict__ W_hh,
                       __bf16* __restrict__ Wp) {
    size_t idx = (size_t)blockIdx.x * blockDim.x + threadIdx.x;
    if (idx >= (size_t)NTOT * KTOT) return;
    int e    = (int)(idx & 15);
    int lane = (int)((idx >> 4) & 31);
    size_t tile = idx >> 9;
    int kt = (int)(tile % KTILES);
    int nt = (int)(tile / KTILES);
    int col = lane & 15;
    int k   = (e & 7) + ((e >> 3) << 4) + ((lane >> 4) << 3);
    int n   = nt * 16 + col;
    int K   = kt * 32 + k;
    int r   = (n & 3) * DD + (n >> 2);             // original row of W_ih/W_hh
    float v = (K < 2 * DD) ? W_ih[(size_t)r * (2 * DD) + K]
                           : W_hh[(size_t)r * DD + (K - 2 * DD)];
    Wp[idx] = f2bf(v);
}

__global__ void pack_bias(const float* __restrict__ b_ih, const float* __restrict__ b_hh,
                          float* __restrict__ biasp) {
    int n = blockIdx.x * blockDim.x + threadIdx.x;
    if (n < NTOT) {
        int r = (n & 3) * DD + (n >> 2);
        biasp[n] = b_ih[r] + b_hh[r];
    }
}

__global__ void conv_x(const float* __restrict__ inp, __bf16* __restrict__ xin) {
    size_t i = (size_t)blockIdx.x * blockDim.x + threadIdx.x;
    if (i < (size_t)BD * TT * DD) xin[i] = f2bf(inp[i]);
}

// ---------------------------------------------------------------------------
// Persistent recurrence kernel: 64 blocks x 256 threads = 512 waves.
// Phase 1 (all blocks): gates GEMM (1 tile/wave, 96 bf16 WMMAs) + fused LSTM.
// Phase 2 (blocks 0..31, one per batch): phi, GMM attention weights w,
//   alignment output, ctx = w @ memory, termination/ctx outputs at t=T-1.
// ---------------------------------------------------------------------------

__global__ __launch_bounds__(BTHR) void tts_main(
    const __bf16* __restrict__ xin, const __bf16* __restrict__ Wp,
    const float*  __restrict__ biasp,
    const float*  __restrict__ mem, const int* __restrict__ lens,
    const float*  __restrict__ Wg,  const float* __restrict__ bg,
    float* __restrict__ h, float* __restrict__ c,
    __bf16* __restrict__ hb, __bf16* __restrict__ ctxb,
    float* __restrict__ ksi,
    float* __restrict__ out_ctx, float* __restrict__ out_align,
    float* __restrict__ out_term,
    unsigned* __restrict__ bar) {

    __shared__ float sh_tile[8 * 256];   // per-wave 16x16 C tiles
    __shared__ float sh_red[256];
    __shared__ float sh_w[256];
    __shared__ float sh_phi[32];
    __shared__ float sh_ksi[MM];
    __shared__ float sh_beta[MM];
    __shared__ float sh_alpha[MM];

    const int tid  = threadIdx.x;
    const int lane = tid & 31;
    const int wid  = tid >> 5;
    const int tile = blockIdx.x * 8 + wid;    // 0..511
    const int mt   = tile >> 8;               // batch tile (0..1)
    const int nt   = tile & 255;              // gate-col tile (0..255)
    const int half = lane >> 4;
    const int mrow = mt * 16 + (lane & 15);   // batch row for A fragment

    const __bf16* wp0 = Wp + (((size_t)nt * KTILES) * 32 + lane) * 16;
    const float bias = biasp[nt * 16 + (lane & 15)];

    for (int t = 0; t < TT; ++t) {
        // ---------------- Phase 1: gates GEMM + LSTM ----------------
        v8f acc = {};
        const __bf16* wp = wp0;
        {   // K segment 0: inp_t
            const __bf16* ar = xin + ((size_t)mrow * TT + t) * DD + 8 * half;
            #pragma unroll 4
            for (int kt = 0; kt < 32; ++kt) {
                __builtin_prefetch((const void*)(wp + 2048), 0, 0);
                acc = mma32(ar + kt * 32, wp, acc);
                wp += 512;
            }
        }
        {   // K segment 1: ctx
            const __bf16* ar = ctxb + (size_t)mrow * DD + 8 * half;
            #pragma unroll 4
            for (int kt = 0; kt < 32; ++kt) {
                __builtin_prefetch((const void*)(wp + 2048), 0, 0);
                acc = mma32(ar + kt * 32, wp, acc);
                wp += 512;
            }
        }
        {   // K segment 2: h
            const __bf16* ar = hb + (size_t)mrow * DD + 8 * half;
            #pragma unroll 4
            for (int kt = 0; kt < 32; ++kt) {
                __builtin_prefetch((const void*)(wp + 2048), 0, 0);
                acc = mma32(ar + kt * 32, wp, acc);
                wp += 512;
            }
        }

        // scatter C tile to LDS: row = vgpr + 8*half (batch-in-tile), col = lane&15
        float* tl = sh_tile + wid * 256;
        #pragma unroll
        for (int j = 0; j < 8; ++j)
            tl[(j + 8 * half) * 16 + (lane & 15)] = acc[j] + bias;
        __syncthreads();

        // fused LSTM cell update: 16 cols = 4 channels x 4 gates (i,f,g,o)
        #pragma unroll
        for (int r = 0; r < 2; ++r) {
            int o  = lane + 32 * r;              // 0..63
            int bl = o & 15, dl = o >> 4;
            float ig = tl[bl * 16 + 4 * dl + 0];
            float fg = tl[bl * 16 + 4 * dl + 1];
            float gg = tl[bl * 16 + 4 * dl + 2];
            float og = tl[bl * 16 + 4 * dl + 3];
            int b = mt * 16 + bl;
            int d = nt * 4 + dl;
            size_t off = (size_t)b * DD + d;
            float cn = sigf(fg) * c[off] + sigf(ig) * tanhf(gg);
            float hn = sigf(og) * tanhf(cn);
            c[off] = cn;
            h[off] = hn;
            hb[off] = f2bf(hn);
        }

        grid_barrier(bar);

        // ---------------- Phase 2: attention (block b = blockIdx.x < 32) ----
        if (blockIdx.x < BD) {
            const int b = blockIdx.x;
            const float* hrow = h + (size_t)b * DD;

            // phi = h @ Wg^T + bg   (30 rows)
            for (int r = 0; r < 3 * MM; ++r) {
                const float* wr = Wg + (size_t)r * DD;
                float p = 0.f;
                for (int k = tid; k < DD; k += BTHR) p += hrow[k] * wr[k];
                sh_red[tid] = p; __syncthreads();
                for (int s = 128; s > 0; s >>= 1) {
                    if (tid < s) sh_red[tid] += sh_red[tid + s];
                    __syncthreads();
                }
                if (tid == 0) sh_phi[r] = sh_red[0] + bg[r];
                __syncthreads();
            }

            if (tid < MM) {
                float kn = fminf(ksi[b * MM + tid] + __expf(sh_phi[tid]), (float)SS - 1.0f);
                ksi[b * MM + tid] = kn;
                sh_ksi[tid]  = kn;
                sh_beta[tid] = __expf(sh_phi[MM + tid]);
            }
            __syncthreads();
            if (tid == 0) {   // softmax over 10 mixture logits
                float mx = sh_phi[2 * MM];
                for (int m = 1; m < MM; ++m) mx = fmaxf(mx, sh_phi[2 * MM + m]);
                float e[MM]; float sum = 0.f;
                for (int m = 0; m < MM; ++m) { e[m] = __expf(sh_phi[2 * MM + m] - mx); sum += e[m]; }
                float inv = 1.f / sum;
                for (int m = 0; m < MM; ++m) sh_alpha[m] = e[m] * inv;
            }
            __syncthreads();

            // attention weights (tid == position s)
            float t1 = 0.f, t2 = 0.f;
            #pragma unroll
            for (int m = 0; m < MM; ++m) {
                float inv = 1.0f / sh_beta[m];
                t1 += sh_alpha[m] * sigf(((float)tid + 0.5f - sh_ksi[m]) * inv);
                t2 += sh_alpha[m] * sigf(((float)tid - 0.5f - sh_ksi[m]) * inv);
            }
            float wv = t1 - t2;
            sh_red[tid] = wv; __syncthreads();
            for (int s = 128; s > 0; s >>= 1) {
                if (tid < s) sh_red[tid] += sh_red[tid + s];
                __syncthreads();
            }
            float wn = wv / sh_red[0];
            sh_w[tid] = wn;
            out_align[((size_t)b * TT + t) * SS + tid] = wn;
            if (t == TT - 1) {
                int idx = lens[b] - 1;
                idx = idx < 0 ? 0 : (idx > SS - 1 ? SS - 1 : idx);
                if (tid == idx) out_term[b] = 1.0f - t1;   // pre-normalization term1
            }
            __syncthreads();

            // ctx = w @ memory[b]  (4 outputs per thread)
            #pragma unroll
            for (int kk = 0; kk < 4; ++kk) {
                int d = tid + kk * BTHR;
                const float* mb = mem + ((size_t)b * SS * DD) + d;
                float a2 = 0.f;
                #pragma unroll 4
                for (int s = 0; s < SS; ++s)
                    a2 = fmaf(sh_w[s], mb[(size_t)s * DD], a2);
                ctxb[(size_t)b * DD + d] = f2bf(a2);
                if (t == TT - 1) out_ctx[(size_t)b * DD + d] = a2;
            }
        }

        grid_barrier(bar);
    }
}

// ---------------------------------------------------------------------------
// Host launcher
// ---------------------------------------------------------------------------

extern "C" void kernel_launch(void* const* d_in, const int* in_sizes, int n_in,
                              void* d_out, int out_size, void* d_ws, size_t ws_size,
                              hipStream_t stream) {
    const float* inp  = (const float*)d_in[0];   // [B,T,D]
    const float* mem  = (const float*)d_in[1];   // [B,S,D]
    const int*   lens = (const int*)d_in[2];     // [B]
    const float* W_ih = (const float*)d_in[3];   // [4D, 2D]
    const float* W_hh = (const float*)d_in[4];   // [4D, D]
    const float* b_ih = (const float*)d_in[5];
    const float* b_hh = (const float*)d_in[6];
    const float* W_g  = (const float*)d_in[7];   // [3M, D]
    const float* b_g  = (const float*)d_in[8];

    char* ws = (char*)d_ws;
    size_t off = 0;
    auto take = [&](size_t bytes) -> char* {
        char* p = ws + off;
        off = (off + bytes + 255) & ~(size_t)255;
        return p;
    };
    unsigned* bar  = (unsigned*)take(256);
    float*  c      = (float*)take((size_t)BD * DD * 4);
    float*  h      = (float*)take((size_t)BD * DD * 4);
    __bf16* hb     = (__bf16*)take((size_t)BD * DD * 2);
    __bf16* ctxb   = (__bf16*)take((size_t)BD * DD * 2);
    float*  ksi    = (float*)take((size_t)BD * MM * 4);
    float*  biasp  = (float*)take((size_t)NTOT * 4);
    __bf16* Wp     = (__bf16*)take((size_t)NTOT * KTOT * 2);
    __bf16* xin    = (__bf16*)take((size_t)BD * TT * DD * 2);

    float* out      = (float*)d_out;
    float* out_ctx  = out;                                   // [B,1,D]
    float* out_algn = out + (size_t)BD * DD;                 // [B,T,S]
    float* out_term = out + (size_t)BD * DD + (size_t)BD * TT * SS;  // [B,1]

    init_state<<<(BD * DD + BTHR - 1) / BTHR, BTHR, 0, stream>>>(c, h, hb, ctxb, ksi, bar);

    size_t wtot = (size_t)NTOT * KTOT;
    pack_w<<<(unsigned)((wtot + BTHR - 1) / BTHR), BTHR, 0, stream>>>(W_ih, W_hh, Wp);
    pack_bias<<<(NTOT + BTHR - 1) / BTHR, BTHR, 0, stream>>>(b_ih, b_hh, biasp);

    size_t xtot = (size_t)BD * TT * DD;
    conv_x<<<(unsigned)((xtot + BTHR - 1) / BTHR), BTHR, 0, stream>>>(inp, xin);

    tts_main<<<NBLK, BTHR, 0, stream>>>(xin, Wp, biasp, mem, lens, W_g, b_g,
                                        h, c, hb, ctxb, ksi,
                                        out_ctx, out_algn, out_term, bar);
}
